// E75MultiHead_11416023073344
// MI455X (gfx1250) — matrix operand order repaired and checked by reading, not compiled, not run on hardware
//
#include <hip/hip_runtime.h>
#include <math.h>

// ---------------------------------------------------------------------------
// E75 MultiHead delta-rule cell for MI455X (gfx1250), wave32 + WMMA.
//   split(x,W*) -> inproj(GEMM+silu, writes f16 hi/lo planes)
//   -> proj(GEMM -> k/v/q/beta f32, sigmoid for beta)
//   -> row-parallel scan (writes cell f16 hi/lo)
//   -> outproj(GEMM -> y f32)
// All GEMMs: split-f16 WMMA (hi*hi + hi*lo + lo*hi) for ~fp32 accuracy on the
// f16 XDL pipe; operands are PRE-SPLIT into f16 planes so the inner loop is
// pure b128 loads + v_wmma (no per-chunk cvt VALU work).
// ---------------------------------------------------------------------------

typedef __attribute__((ext_vector_type(16))) _Float16 v16h;
typedef __attribute__((ext_vector_type(8)))  _Float16 v8h;
typedef __attribute__((ext_vector_type(4)))  _Float16 v4h;
typedef __attribute__((ext_vector_type(8)))  float    v8f;

#define B_   8
#define T_   2048
#define D_   1024
#define H_   8
#define N_   32
#define M_   (B_ * T_)     // 16384 rows
#define HN_  (H_ * N_)     // 256

// ----------------------------- helpers -------------------------------------

__device__ __forceinline__ float wave_sum32(float v) {
#pragma unroll
  for (int m = 16; m >= 1; m >>= 1) v += __shfl_xor(v, m, 32);
  return v;
}

__device__ __forceinline__ v16h cat8(v8h a, v8h b) {
  return __builtin_shufflevector(a, b, 0, 1, 2, 3, 4, 5, 6, 7,
                                 8, 9, 10, 11, 12, 13, 14, 15);
}

// 16x64 output tile per wave (4 x 16x16 accumulators).
//   out[m, n] = sum_k A[m,k] * W[n,k]
// A/W given as pre-split f16 hi/lo planes, row-major with leading dim K.
// Per-lane layout (this code's contiguous k order, same for A and B):
//   A: lane L holds row L%16, halves 0-15 = K kb+g*16..  (g = L/16)
//   B: lane L holds W-row (out col) L%16, halves 0-15 = K kb+g*16..  (g = L/16)
__device__ __forceinline__ void wmma_tile_16x64(
    const _Float16* __restrict__ Ahi, const _Float16* __restrict__ Alo,
    const _Float16* __restrict__ Whi, const _Float16* __restrict__ Wlo,
    int K, int lane, v8f acc[4]) {
  const int mr  = lane & 15;
  const int grp = lane >> 4;
  const _Float16* ah0 = Ahi + (size_t)mr * K + grp * 16;
  const _Float16* al0 = Alo + (size_t)mr * K + grp * 16;
  const size_t wofs = (size_t)mr * K + grp * 16;

  for (int kb = 0; kb < K; kb += 32) {
    const v16h ah = cat8(*(const v8h*)(ah0 + kb), *(const v8h*)(ah0 + kb + 8));
    const v16h al = cat8(*(const v8h*)(al0 + kb), *(const v8h*)(al0 + kb + 8));
#pragma unroll
    for (int s = 0; s < 4; ++s) {
      const _Float16* wh = Whi + (size_t)(s * 16) * K + wofs + kb;
      const _Float16* wl = Wlo + (size_t)(s * 16) * K + wofs + kb;
      const v16h bh = cat8(*(const v8h*)wh, *(const v8h*)(wh + 8));
      const v16h bl = cat8(*(const v8h*)wl, *(const v8h*)(wl + 8));
      acc[s] = __builtin_amdgcn_wmma_f32_16x16x32_f16(false, al, false, bh, (short)0, acc[s], false, false);
      acc[s] = __builtin_amdgcn_wmma_f32_16x16x32_f16(false, ah, false, bl, (short)0, acc[s], false, false);
      acc[s] = __builtin_amdgcn_wmma_f32_16x16x32_f16(false, ah, false, bh, (short)0, acc[s], false, false);
    }
  }
}

// --------------------- kernel 0: fp32 -> f16 hi/lo split --------------------

__global__ void e75_split_kernel(const float* __restrict__ src,
                                 _Float16* __restrict__ hi,
                                 _Float16* __restrict__ lo, int count4) {
  const int idx = blockIdx.x * blockDim.x + threadIdx.x;
  if (idx >= count4) return;
  const float4 f = ((const float4*)src)[idx];
  const _Float16 h0 = (_Float16)f.x, h1 = (_Float16)f.y;
  const _Float16 h2 = (_Float16)f.z, h3 = (_Float16)f.w;
  v4h hv = {h0, h1, h2, h3};
  v4h lv = {(_Float16)(f.x - (float)h0), (_Float16)(f.y - (float)h1),
            (_Float16)(f.z - (float)h2), (_Float16)(f.w - (float)h3)};
  *(v4h*)(hi + (size_t)idx * 4) = hv;
  *(v4h*)(lo + (size_t)idx * 4) = lv;
}

// ----------------------------- kernel 1: in_proj + SiLU --------------------

__global__ void e75_inproj_kernel(const _Float16* __restrict__ xhi,
                                  const _Float16* __restrict__ xlo,
                                  const _Float16* __restrict__ winhi,
                                  const _Float16* __restrict__ winlo,
                                  _Float16* __restrict__ xphi,
                                  _Float16* __restrict__ xplo) {
  const int wave = (blockIdx.x * blockDim.x + threadIdx.x) >> 5;
  const int lane = threadIdx.x & 31;
  const int nt   = D_ / 64;                    // 16 col tiles of width 64
  const int tm   = (wave / nt) * 16;
  const int tn   = (wave % nt) * 64;

  v8f acc[4] = {};
  wmma_tile_16x64(xhi + (size_t)tm * D_, xlo + (size_t)tm * D_,
                  winhi + (size_t)tn * D_, winlo + (size_t)tn * D_, D_, lane, acc);

#pragma unroll
  for (int s = 0; s < 4; ++s) {
    const int n = tn + s * 16 + (lane & 15);
#pragma unroll
    for (int r = 0; r < 8; ++r) {
      const int m = tm + r + (lane >> 4) * 8;
      float v = acc[s][r];
      v = v / (1.0f + __expf(-v));             // SiLU
      const _Float16 h = (_Float16)v;
      xphi[(size_t)m * D_ + n] = h;
      xplo[(size_t)m * D_ + n] = (_Float16)(v - (float)h);
    }
  }
}

// ------------------- kernel 2: fused k/v/q/beta projections ----------------
// Output layout per tensor: f32 [B,H,T,N] (contiguous 32-float vectors per
// (b,h,t) for the scan). Beta gets sigmoid(val + b_beta[h,n]).

__global__ void e75_proj_kernel(const _Float16* __restrict__ xphi,
                                const _Float16* __restrict__ xplo,
                                const _Float16* __restrict__ wkhi, const _Float16* __restrict__ wklo,
                                const _Float16* __restrict__ wvhi, const _Float16* __restrict__ wvlo,
                                const _Float16* __restrict__ wqhi, const _Float16* __restrict__ wqlo,
                                const _Float16* __restrict__ wbhi, const _Float16* __restrict__ wblo,
                                const float* __restrict__ bbeta,
                                float* __restrict__ kx, float* __restrict__ vx,
                                float* __restrict__ qx, float* __restrict__ betax) {
  const int wave = (blockIdx.x * blockDim.x + threadIdx.x) >> 5;
  const int lane = threadIdx.x & 31;
  const int nt   = (4 * HN_) / 64;             // 16 tiles over concat(4 x 256)
  const int tm   = (wave / nt) * 16;
  const int tcol = (wave % nt) * 64;           // 0..960
  const int wsel = tcol >> 8;                  // tile of 64 never crosses a weight
  const int c0   = tcol & 255;

  const _Float16* Whi = (wsel == 0) ? wkhi : (wsel == 1) ? wvhi : (wsel == 2) ? wqhi : wbhi;
  const _Float16* Wlo = (wsel == 0) ? wklo : (wsel == 1) ? wvlo : (wsel == 2) ? wqlo : wblo;
  float* dst = (wsel == 0) ? kx : (wsel == 1) ? vx : (wsel == 2) ? qx : betax;

  v8f acc[4] = {};
  wmma_tile_16x64(xphi + (size_t)tm * D_, xplo + (size_t)tm * D_,
                  Whi + (size_t)c0 * D_, Wlo + (size_t)c0 * D_, D_, lane, acc);

#pragma unroll
  for (int s = 0; s < 4; ++s) {
    const int c   = c0 + s * 16 + (lane & 15); // 0..255
    const int h   = c >> 5;
    const int n32 = c & 31;
    const float bias = (wsel == 3) ? bbeta[h * N_ + n32] : 0.0f;
#pragma unroll
    for (int r = 0; r < 8; ++r) {
      const int m = tm + r + (lane >> 4) * 8;
      const int b = m >> 11;                   // m / T_
      const int t = m & (T_ - 1);
      float v = acc[s][r];
      if (wsel == 3) v = 1.0f / (1.0f + __expf(-(v + bias)));
      dst[((size_t)(b * H_ + h) * T_ + t) * N_ + n32] = v;
    }
  }
}

// --------------------------- kernel 3: recurrent scan ----------------------
// One wave32 per (b,h,row): lane j owns S[i][j]. Cross-lane sums are shfl_xor
// butterflies. 2048 independent waves. Cell output written pre-split f16.

__global__ void e75_scan_kernel(const float* __restrict__ kx,
                                const float* __restrict__ vx,
                                const float* __restrict__ qx,
                                const float* __restrict__ betax,
                                _Float16* __restrict__ cellhi,
                                _Float16* __restrict__ celllo,
                                float* __restrict__ s_final) {
  const int wave = (blockIdx.x * blockDim.x + threadIdx.x) >> 5;
  const int lane = threadIdx.x & 31;           // j (column of S)
  const int bh   = wave >> 5;                  // 0..63  (b*H + h)
  const int i    = wave & 31;                  // row of S
  const int b    = bh >> 3;
  const int h    = bh & 7;

  const float* kb = kx    + (size_t)bh * T_ * N_;
  const float* vb = vx    + (size_t)bh * T_ * N_;
  const float* qb = qx    + (size_t)bh * T_ * N_;
  const float* bb = betax + (size_t)bh * T_ * N_;

  float S = 0.0f;                               // S[i][lane]

  for (int t = 0; t < T_; ++t) {
    const float kj = kb[t * N_ + lane];
    const float qj = qb[t * N_ + lane];
    const float vi = vb[t * N_ + i];            // broadcast load
    const float bi = bb[t * N_ + i];            // broadcast load

    const float ss  = wave_sum32(kj * kj);
    const float inv = 1.0f / (sqrtf(ss) + 1e-6f);
    const float kn  = kj * inv;

    const float retr  = wave_sum32(S * kn);
    const float delta = vi - retr;

    S = tanhf(bi * S + delta * kn);

    const float Sq = wave_sum32(S * qj);
    if (lane == 0) {
      const float sig = 1.0f / (1.0f + __expf(-Sq));
      const float o   = Sq * Sq * sig;          // Sq * silu(Sq)
      const size_t idx = (size_t)(b * T_ + t) * HN_ + h * N_ + i;
      const _Float16 oh = (_Float16)o;
      cellhi[idx] = oh;
      celllo[idx] = (_Float16)(o - (float)oh);
    }
  }
  // S_final [B,H,N,N]
  s_final[((size_t)bh * N_ + i) * N_ + lane] = S;
}

// --------------------------- kernel 4: out_proj ----------------------------

__global__ void e75_outproj_kernel(const _Float16* __restrict__ cellhi,
                                   const _Float16* __restrict__ celllo,
                                   const _Float16* __restrict__ wouthi,
                                   const _Float16* __restrict__ woutlo,
                                   float* __restrict__ y) {
  const int wave = (blockIdx.x * blockDim.x + threadIdx.x) >> 5;
  const int lane = threadIdx.x & 31;
  const int nt   = D_ / 64;
  const int tm   = (wave / nt) * 16;
  const int tn   = (wave % nt) * 64;

  v8f acc[4] = {};
  wmma_tile_16x64(cellhi + (size_t)tm * HN_, celllo + (size_t)tm * HN_,
                  wouthi + (size_t)tn * HN_, woutlo + (size_t)tn * HN_, HN_, lane, acc);

#pragma unroll
  for (int s = 0; s < 4; ++s) {
    const int n = tn + s * 16 + (lane & 15);
#pragma unroll
    for (int r = 0; r < 8; ++r) {
      const int m = tm + r + (lane >> 4) * 8;
      y[(size_t)m * D_ + n] = acc[s][r];
    }
  }
}

// ------------------------------- launcher ----------------------------------

extern "C" void kernel_launch(void* const* d_in, const int* in_sizes, int n_in,
                              void* d_out, int out_size, void* d_ws, size_t ws_size,
                              hipStream_t stream) {
  (void)in_sizes; (void)n_in; (void)out_size; (void)ws_size;

  const float* x      = (const float*)d_in[0];
  const float* W_in   = (const float*)d_in[1];
  const float* W_k    = (const float*)d_in[2];
  const float* W_v    = (const float*)d_in[3];
  const float* W_q    = (const float*)d_in[4];
  const float* W_beta = (const float*)d_in[5];
  const float* b_beta = (const float*)d_in[6];
  const float* W_out  = (const float*)d_in[7];

  float* y       = (float*)d_out;                    // [B,T,D]
  float* s_final = y + (size_t)M_ * D_;              // [B,H,N,N]

  // ---- workspace carve-up ----
  char* p = (char*)d_ws;
  auto alloc_f = [&](size_t n) { float* r = (float*)p;     p += n * sizeof(float);     return r; };
  auto alloc_h = [&](size_t n) { _Float16* r = (_Float16*)p; p += n * sizeof(_Float16); return r; };

  float*    kx     = alloc_f((size_t)B_ * H_ * T_ * N_);   // 16 MB
  float*    vx     = alloc_f((size_t)B_ * H_ * T_ * N_);
  float*    qx     = alloc_f((size_t)B_ * H_ * T_ * N_);
  float*    betax  = alloc_f((size_t)B_ * H_ * T_ * N_);
  _Float16* xhi    = alloc_h((size_t)M_ * D_);             // 33.5 MB each
  _Float16* xlo    = alloc_h((size_t)M_ * D_);
  _Float16* xphi   = alloc_h((size_t)M_ * D_);
  _Float16* xplo   = alloc_h((size_t)M_ * D_);
  _Float16* cellhi = alloc_h((size_t)M_ * HN_);            // 8.4 MB each
  _Float16* celllo = alloc_h((size_t)M_ * HN_);
  _Float16* winhi  = alloc_h((size_t)D_ * D_);
  _Float16* winlo  = alloc_h((size_t)D_ * D_);
  _Float16* wkhi   = alloc_h((size_t)HN_ * D_);
  _Float16* wklo   = alloc_h((size_t)HN_ * D_);
  _Float16* wvhi   = alloc_h((size_t)HN_ * D_);
  _Float16* wvlo   = alloc_h((size_t)HN_ * D_);
  _Float16* wqhi   = alloc_h((size_t)HN_ * D_);
  _Float16* wqlo   = alloc_h((size_t)HN_ * D_);
  _Float16* wbhi   = alloc_h((size_t)HN_ * D_);
  _Float16* wblo   = alloc_h((size_t)HN_ * D_);
  _Float16* wohi   = alloc_h((size_t)D_ * HN_);
  _Float16* wolo   = alloc_h((size_t)D_ * HN_);

  const dim3 blk(256);                               // 8 waves per block

  // 0) one-time operand splits (fp32 -> f16 hi/lo planes)
  auto split = [&](const float* s, _Float16* hi, _Float16* lo, size_t count) {
    const int c4 = (int)(count / 4);
    e75_split_kernel<<<dim3((c4 + 255) / 256), blk, 0, stream>>>(s, hi, lo, c4);
  };
  split(x,      xhi,   xlo,   (size_t)M_ * D_);
  split(W_in,   winhi, winlo, (size_t)D_ * D_);
  split(W_k,    wkhi,  wklo,  (size_t)HN_ * D_);
  split(W_v,    wvhi,  wvlo,  (size_t)HN_ * D_);
  split(W_q,    wqhi,  wqlo,  (size_t)HN_ * D_);
  split(W_beta, wbhi,  wblo,  (size_t)HN_ * D_);
  split(W_out,  wohi,  wolo,  (size_t)D_ * HN_);

  // 1) in_proj + SiLU: (M/16)*(D/64) = 16384 waves -> 2048 blocks
  e75_inproj_kernel<<<dim3((M_ / 16) * (D_ / 64) / 8), blk, 0, stream>>>(
      xhi, xlo, winhi, winlo, xphi, xplo);

  // 2) fused 4-way projection: 16384 waves
  e75_proj_kernel<<<dim3((M_ / 16) * ((4 * HN_) / 64) / 8), blk, 0, stream>>>(
      xphi, xplo, wkhi, wklo, wvhi, wvlo, wqhi, wqlo, wbhi, wblo,
      b_beta, kx, vx, qx, betax);

  // 3) scan: B*H*N = 2048 waves -> 256 blocks
  e75_scan_kernel<<<dim3(B_ * H_ * N_ / 8), blk, 0, stream>>>(
      kx, vx, qx, betax, cellhi, celllo, s_final);

  // 4) out_proj: 16384 waves
  e75_outproj_kernel<<<dim3((M_ / 16) * (D_ / 64) / 8), blk, 0, stream>>>(
      cellhi, celllo, wohi, wolo, y);
}
